// GAFlowCore_78932908966242
// MI455X (gfx1250) — matrix operand order repaired and checked by reading, not compiled, not run on hardware
//
#include <hip/hip_runtime.h>
#include <hip/hip_bf16.h>

// RAFT correlation pyramid for MI455X (gfx1250, wave32, WMMA + TDM).
//   pack_a : f1 [256][8192] f32 -> bf16 WMMA A-fragment order  (4 MB)
//   pack_b : f2 pooled per level -> bf16 WMMA B-fragment order (5.6 MB)
//   gemm16 : CV_i[m,n] = <f1_m, pool_i(f2)_n>/16. B tile (32 KB, contiguous in
//            packed layout) staged once per block into LDS by the Tensor Data
//            Mover (TENSORcnt), consumed by 8 waves via ds_load_b128;
//            32x v_wmma_f32_16x16x32_bf16 per wave. M chunked so the CV slab
//            (44.6 MB) stays resident in the 192 MB L2.
//   sampler: 4*81 bilinear lookups per pixel from the L2-resident slab.
// Workspace use: ~54.2 MB.

typedef __attribute__((ext_vector_type(16))) __bf16 v16bf;
typedef __attribute__((ext_vector_type(8)))  __bf16 v8bf;
typedef __attribute__((ext_vector_type(8)))  float  v8f;
typedef __attribute__((ext_vector_type(4)))  unsigned int u32x4;
typedef __attribute__((ext_vector_type(8)))  int    i32x8;
typedef __attribute__((ext_vector_type(4)))  int    i32x4;

#define H0      64
#define W0      128
#define HW0     8192
#define DIM     256
#define CHUNK_M 1024

// LDS chunk stride: 32B payload + 16B pad -> bank stride 12 dwords (2-way max)
#define LDS_STRIDE 48
#define LDS_BYTES  (4 * 8 * 32 * LDS_STRIDE)   // 49152

static __device__ __forceinline__ __bf16 f2bf(float f) {
  unsigned u = __builtin_bit_cast(unsigned, f);
  unsigned r = (u + 0x7fffu + ((u >> 16) & 1u)) >> 16;   // round-nearest-even
  return __builtin_bit_cast(__bf16, (unsigned short)r);
}

static __device__ __forceinline__ unsigned lds_byte_offset(const void* p) {
  // generic -> LDS(AS3) addrspacecast; AS3 pointers are raw LDS byte offsets
  return (unsigned)(unsigned long long)
         (__attribute__((address_space(3))) const char*)(const char*)p;
}

// ---- pack fmap1 into WMMA 16-bit A-matrix (16x32) fragment order ----------
// Per ISA 7.12.2: lanes 0-15 hold K {0..7,16..23}, lanes 16-31 hold {8..15,24..31}.
__global__ void __launch_bounds__(256)
pack_a_kernel(const float* __restrict__ f, __bf16* __restrict__ out) {
  int tid  = blockIdx.x * 256 + threadIdx.x;
  int j    = tid & 15;
  int lane = (tid >> 4) & 31;
  int ks   = (tid >> 9) & 7;
  int mt   = tid >> 12;
  int m    = mt * 16 + (lane & 15);
  int kb   = (lane >> 4) * 8;
  int c    = ks * 32 + ((j < 8) ? (kb + j) : (kb + 16 + (j - 8)));
  out[tid] = f2bf(f[c * HW0 + m]);
}

// ---- pool fmap2 (level i) and pack into B-matrix (32x16) fragment order ---
// Per-lane B: n = lane%16, k = 8*(lane/16) + j (j<8), 16 + 8*(lane/16) + (j-8) (j>=8).
__global__ void __launch_bounds__(256)
pack_b_kernel(const float* __restrict__ f, __bf16* __restrict__ out,
              int Wi, int pool) {
  int tid  = blockIdx.x * 256 + threadIdx.x;
  int j    = tid & 15;
  int lane = (tid >> 4) & 31;
  int ks   = (tid >> 9) & 7;
  int nt   = tid >> 12;
  int n    = nt * 16 + (lane & 15);
  int c    = ks * 32 + (lane >> 4) * 8 + ((j < 8) ? j : (16 + (j - 8)));
  int ny   = n / Wi, nx = n % Wi;
  float s = 0.f;
  for (int py = 0; py < pool; ++py)
    for (int px = 0; px < pool; ++px)
      s += f[c * HW0 + (ny * pool + py) * W0 + (nx * pool + px)];
  out[tid] = f2bf(s * (1.0f / (float)(pool * pool)));
}

// ---- read one 16x32 bf16 fragment (32B/lane) from padded LDS --------------
static __device__ __forceinline__ v16bf lds_frag(const char* bs, int chunk) {
  const v8bf* p = (const v8bf*)(bs + (size_t)chunk * LDS_STRIDE);
  v8bf h0 = p[0];
  v8bf h1 = p[1];
  v16bf r;
#pragma unroll
  for (int i = 0; i < 8; ++i) { r[i] = h0[i]; r[i + 8] = h1[i]; }
  return r;
}

// ---- bf16 WMMA GEMM: C[mlocal, n] = scale * sum_c A[m,c] B[c,n] -----------
// Block = 8 waves, one 16-row m-tile each, sharing one 64-col n-group whose
// 32 KB of packed B fragments are DMA'd into LDS once by the TDM.
__global__ void __launch_bounds__(256)
gemm16_kernel(const __bf16* __restrict__ A, const __bf16* __restrict__ B,
              float* __restrict__ C, int mtOff, int N, float scale) {
  __shared__ __align__(16) char bs[LDS_BYTES];

  int lane = threadIdx.x & 31;
  int wave = threadIdx.x >> 5;
  int mtl  = blockIdx.y * 8 + wave;   // chunk-local m-tile
  int mt   = mtOff + mtl;             // global m-tile
  int g    = blockIdx.x;              // group of 4 n-tiles

#if __has_builtin(__builtin_amdgcn_tensor_load_to_lds)
  if (threadIdx.x < 32) {             // wave 0 drives the Tensor Data Mover
    unsigned long long ga =
        (unsigned long long)(const void*)B + (unsigned long long)g * 32768ull;
    unsigned lo = lds_byte_offset(bs);
    u32x4 g0;
    g0[0] = 1u;                                        // count=1 (valid D#)
    g0[1] = lo;                                        // lds_addr
    g0[2] = (unsigned)(ga & 0xFFFFFFFFu);              // global_addr[31:0]
    g0[3] = (unsigned)((ga >> 32) & 0x01FFFFFFu)       // global_addr[56:32]
            | 0x80000000u;                             // type=2 ("image")
    i32x8 g1;
    g1[0] = (int)((1u << 16)        // data_size = 2 bytes
                | (1u << 20)        // pad_enable
                | (2u << 22)        // pad_interval: every 8 DWORDs (32B)
                | (3u << 25));      // pad_amount:   4 DWORDs (16B)
    g1[1] = (int)0x40000000;        // tensor_dim0 = 16384 elems (bits 79:48)
    g1[2] = (int)0x00010000;        // tensor_dim1 = 1          (bits 111:80)
    g1[3] = (int)0x40000000;        // tile_dim0   = 16384      (bits 127:112)
    g1[4] = 1;                      // tile_dim1 = 1, tile_dim2 = 0
    g1[5] = 16384;                  // tensor_dim0_stride (bits 207:160)
    g1[6] = 0;
    g1[7] = 0;
    i32x4 z = {0, 0, 0, 0};
#if __clang_major__ >= 23
    i32x8 z8 = {0, 0, 0, 0, 0, 0, 0, 0};
    __builtin_amdgcn_tensor_load_to_lds(g0, g1, z, z, z8, 0);
#else
    __builtin_amdgcn_tensor_load_to_lds(g0, g1, z, z, 0);
#endif
    __builtin_amdgcn_s_wait_tensorcnt(0);
  }
#else
  // Fallback: cooperative copy, 1024 chunks of 32B into the padded layout.
  for (int ch = threadIdx.x; ch < 1024; ch += 256) {
    const v8bf* src = (const v8bf*)(B + (size_t)g * 16384 + (size_t)ch * 16);
    v8bf* dst = (v8bf*)(bs + (size_t)ch * LDS_STRIDE);
    dst[0] = src[0];
    dst[1] = src[1];
  }
#endif
  __syncthreads();

  const v16bf* Af = (const v16bf*)A;
  v8f acc0 = {0.f, 0.f, 0.f, 0.f, 0.f, 0.f, 0.f, 0.f};
  v8f acc1 = acc0, acc2 = acc0, acc3 = acc0;

#pragma unroll
  for (int ks = 0; ks < 8; ++ks) {
    v16bf a  = Af[(mt * 8 + ks) * 32 + lane];
    v16bf b0 = lds_frag(bs, (0 * 8 + ks) * 32 + lane);
    v16bf b1 = lds_frag(bs, (1 * 8 + ks) * 32 + lane);
    v16bf b2 = lds_frag(bs, (2 * 8 + ks) * 32 + lane);
    v16bf b3 = lds_frag(bs, (3 * 8 + ks) * 32 + lane);
    acc0 = __builtin_amdgcn_wmma_f32_16x16x32_bf16(false, a, false, b0,
                                                   (short)0, acc0, false, false);
    acc1 = __builtin_amdgcn_wmma_f32_16x16x32_bf16(false, a, false, b1,
                                                   (short)0, acc1, false, false);
    acc2 = __builtin_amdgcn_wmma_f32_16x16x32_bf16(false, a, false, b2,
                                                   (short)0, acc2, false, false);
    acc3 = __builtin_amdgcn_wmma_f32_16x16x32_bf16(false, a, false, b3,
                                                   (short)0, acc3, false, false);
  }

  // C/D layout: VGPR j -> M = j (lanes 0-15) / j+8 (lanes 16-31), N = lane%16
  int mrow = mtl * 16 + ((lane >> 4) << 3);
  int ncol = g * 64 + (lane & 15);
  float* Cp = C + (size_t)mrow * N + ncol;
#pragma unroll
  for (int jj = 0; jj < 8; ++jj) {
    size_t r = (size_t)jj * N;
    Cp[r +  0] = acc0[jj] * scale;
    Cp[r + 16] = acc1[jj] * scale;
    Cp[r + 32] = acc2[jj] * scale;
    Cp[r + 48] = acc3[jj] * scale;
  }
}

// ---- bilinear lookup sampler (reads L2-resident CV slab) ------------------
__global__ void __launch_bounds__(256)
sampler_kernel(const float* __restrict__ coords,
               const float* __restrict__ cv0, const float* __restrict__ cv1,
               const float* __restrict__ cv2, const float* __restrict__ cv3,
               float* __restrict__ out, int m0) {
  int ml  = blockIdx.x * 256 + threadIdx.x;   // 0..CHUNK_M-1
  int m   = m0 + ml;
  int os  = blockIdx.y;                       // 0..323
  int lvl = os / 81;
  int s   = os % 81;
  // reference quirk: off_x comes from the slow meshgrid axis
  float ox = (float)(s / 9) - 4.0f;
  float oy = (float)(s % 9) - 4.0f;

  float cx  = coords[m];
  float cy  = coords[HW0 + m];
  float inv = 1.0f / (float)(1 << lvl);
  int   Wi  = W0 >> lvl;
  int   Hi  = H0 >> lvl;
  const float* cv  = (lvl == 0) ? cv0 : (lvl == 1) ? cv1 : (lvl == 2) ? cv2 : cv3;
  const float* row = cv + (size_t)ml * (size_t)(Wi * Hi);

  float xs  = cx * inv + ox;
  float ys  = cy * inv + oy;
  float x0f = floorf(xs), y0f = floorf(ys);
  float wx1 = xs - x0f, wx0 = 1.0f - wx1;
  float wy1 = ys - y0f, wy0 = 1.0f - wy1;
  int x0 = (int)x0f, y0 = (int)y0f, x1 = x0 + 1, y1 = y0 + 1;

  auto gat = [&](int yc, int xc) -> float {
    return (xc >= 0 && xc < Wi && yc >= 0 && yc < Hi) ? row[yc * Wi + xc] : 0.0f;
  };
  out[(size_t)os * HW0 + m] = gat(y0, x0) * wy0 * wx0 + gat(y0, x1) * wy0 * wx1 +
                              gat(y1, x0) * wy1 * wx0 + gat(y1, x1) * wy1 * wx1;
}

// ---- host orchestration ---------------------------------------------------
extern "C" void kernel_launch(void* const* d_in, const int* in_sizes, int n_in,
                              void* d_out, int out_size, void* d_ws, size_t ws_size,
                              hipStream_t stream) {
  const float* fmap1  = (const float*)d_in[0];
  const float* fmap2  = (const float*)d_in[1];
  const float* coords = (const float*)d_in[2];
  float* out = (float*)d_out;
  char*  ws  = (char*)d_ws;

  // workspace layout (bytes); total ~54.2 MB (assumed <= ws_size)
  __bf16* A  = (__bf16*)ws;                         // 8192*256*2 = 4 MB
  __bf16* B0 = (__bf16*)(ws + (size_t)4194304);     // 8192*256*2
  __bf16* B1 = B0 + 8192 * 256;                     // 2048*256*2
  __bf16* B2 = B1 + 2048 * 256;                     //  512*256*2
  __bf16* B3 = B2 + 512 * 256;                      //  128*256*2
  float*  CV0 = (float*)(ws + (size_t)9764864);     // CHUNK_M*8192*4
  float*  CV1 = CV0 + (size_t)CHUNK_M * 8192;       // CHUNK_M*2048*4
  float*  CV2 = CV1 + (size_t)CHUNK_M * 2048;       // CHUNK_M*512*4
  float*  CV3 = CV2 + (size_t)CHUNK_M * 512;        // CHUNK_M*128*4

  // operand packing (once)
  pack_a_kernel<<<8192, 256, 0, stream>>>(fmap1, A);
  pack_b_kernel<<<8192, 256, 0, stream>>>(fmap2, B0, 128, 1);
  pack_b_kernel<<<2048, 256, 0, stream>>>(fmap2, B1,  64, 2);
  pack_b_kernel<<< 512, 256, 0, stream>>>(fmap2, B2,  32, 4);
  pack_b_kernel<<< 128, 256, 0, stream>>>(fmap2, B3,  16, 8);

  const float scale = 1.0f / 16.0f;   // 1/sqrt(256); pooling mean is in pack_b
  const int nChunks = HW0 / CHUNK_M;  // 8
  for (int chunk = 0; chunk < nChunks; ++chunk) {
    int mtOff = chunk * (CHUNK_M / 16);
    dim3 blk(256);
    gemm16_kernel<<<dim3(8192 / 64, CHUNK_M / 128), blk, 0, stream>>>(A, B0, CV0, mtOff, 8192, scale);
    gemm16_kernel<<<dim3(2048 / 64, CHUNK_M / 128), blk, 0, stream>>>(A, B1, CV1, mtOff, 2048, scale);
    gemm16_kernel<<<dim3( 512 / 64, CHUNK_M / 128), blk, 0, stream>>>(A, B2, CV2, mtOff,  512, scale);
    gemm16_kernel<<<dim3( 128 / 64, CHUNK_M / 128), blk, 0, stream>>>(A, B3, CV3, mtOff,  128, scale);
    sampler_kernel<<<dim3(CHUNK_M / 256, 324), blk, 0, stream>>>(
        coords, CV0, CV1, CV2, CV3, out, chunk * CHUNK_M);
  }
}
